// Classify_graph_128849019555
// MI455X (gfx1250) — matrix operand order repaired and checked by reading, not compiled, not run on hardware
//
#include <hip/hip_runtime.h>
#include <hip/hip_bf16.h>
#include <math.h>

// ---------------- problem constants (match reference) ----------------
#define N_NODES   100000
#define N_EDGES   800000
#define N_FEAT    128
#define N_HID     128
#define N_LAYERS  3
#define N_CLASSES 10
#define N_GRAPHS  256

#define M_TILES   (N_NODES / 16)     // 6250 (exact)
#define TOT_EDGES (N_EDGES + N_NODES)

typedef __attribute__((ext_vector_type(16))) _Float16 v16h;
typedef __attribute__((ext_vector_type(8)))  float    v8f;

// ---------------- degree / normalization ----------------
__global__ void deg_init_kernel(float* deg) {
    int i = blockIdx.x * blockDim.x + threadIdx.x;
    if (i < N_NODES) deg[i] = 1.0f;            // self-loop
}

__global__ void deg_accum_kernel(const long long* ei, float* deg) {
    int e = blockIdx.x * blockDim.x + threadIdx.x;
    if (e < N_EDGES) {
        int d = (int)ei[(size_t)N_EDGES + e];  // dst row
        atomicAdd(&deg[d], 1.0f);
    }
}

__global__ void deg_finalize_kernel(float* deg) {
    int i = blockIdx.x * blockDim.x + threadIdx.x;
    if (i < N_NODES) deg[i] = rsqrtf(deg[i]);
}

// ---------------- dense transform: H = X @ W via WMMA ----------------
// Block: 128 threads = 4 waves; each wave computes a 16-row strip x 128 cols.
// W (128x128 f32, row-major [k][n]) is staged transposed+converted to f16 in
// LDS as ldsW[n][k] so B-fragments are contiguous 32B loads.
__global__ void gemm_wmma_kernel(const float* __restrict__ X,
                                 const float* __restrict__ W,
                                 float* __restrict__ H) {
    __shared__ _Float16 ldsW[N_HID * N_HID];   // 32 KB

    const int tid = threadIdx.x;
    // stage W: read coalesced over n, write transposed
    for (int i = 0; i < 128; ++i) {
        ldsW[tid * 128 + i] = (_Float16)W[i * 128 + tid];
    }
    __syncthreads();

    const int lane  = tid & 31;
    const int wave  = tid >> 5;
    const int mtile = blockIdx.x * 4 + wave;
    if (mtile >= M_TILES) return;              // uniform per wave

    const int mrow = (lane & 15);
    const int row  = mtile * 16 + mrow;
    const float* xr = X + (size_t)row * 128;

    const int hi   = lane >> 4;                // 0 or 1 (half-wave)
    const int kgrp = hi * 16;                  // A-frag K sub-base
    const int kbB  = hi * 16;                  // B-frag K sub-base
    const int col  = lane & 15;

    v8f acc[8];
    #pragma unroll
    for (int nt = 0; nt < 8; ++nt) acc[nt] = (v8f){0,0,0,0,0,0,0,0};

    #pragma unroll
    for (int kt = 0; kt < 4; ++kt) {
        // ---- A fragment: 16x32 f16, k = 16*hi + s, same order as B ----
        const float* xk = xr + kt * 32 + kgrp;
        float4 p0 = *(const float4*)(xk + 0);
        float4 p1 = *(const float4*)(xk + 4);
        float4 p2 = *(const float4*)(xk + 8);
        float4 p3 = *(const float4*)(xk + 12);
        v16h a;
        a[0]=(_Float16)p0.x;  a[1]=(_Float16)p0.y;  a[2]=(_Float16)p0.z;  a[3]=(_Float16)p0.w;
        a[4]=(_Float16)p1.x;  a[5]=(_Float16)p1.y;  a[6]=(_Float16)p1.z;  a[7]=(_Float16)p1.w;
        a[8]=(_Float16)p2.x;  a[9]=(_Float16)p2.y;  a[10]=(_Float16)p2.z; a[11]=(_Float16)p2.w;
        a[12]=(_Float16)p3.x; a[13]=(_Float16)p3.y; a[14]=(_Float16)p3.z; a[15]=(_Float16)p3.w;

        const int koff = kt * 32 + kbB;
        #pragma unroll
        for (int nt = 0; nt < 8; ++nt) {
            // ---- B fragment: 32x16 f16, contiguous in transposed LDS ----
            v16h b = *(const v16h*)(&ldsW[(nt * 16 + col) * 128 + koff]);
            acc[nt] = __builtin_amdgcn_wmma_f32_16x16x32_f16(
                false, a, false, b, (short)0, acc[nt], false, false);
        }
    }

    // ---- store C/D: lane holds n = lane&15, rows m = r + hi*8 ----
    const int mbase = mtile * 16 + hi * 8;
    #pragma unroll
    for (int nt = 0; nt < 8; ++nt) {
        #pragma unroll
        for (int r = 0; r < 8; ++r) {
            H[(size_t)(mbase + r) * 128 + nt * 16 + col] = acc[nt][r];
        }
    }
}

// ---------------- zero the accumulator buffer ----------------
__global__ void zero_kernel(float4* p, int n4) {
    int i = blockIdx.x * blockDim.x + threadIdx.x;
    if (i < n4) p[i] = (float4){0.f, 0.f, 0.f, 0.f};
}

// ---------------- edge scatter: X[dst] += H[src] * norm ----------------
// one wave per edge; 4 floats per lane (128-wide message)
__global__ void scatter_kernel(const long long* __restrict__ ei,
                               const float* __restrict__ dinv,
                               const float* __restrict__ H,
                               float* __restrict__ X) {
    int gtid = blockIdx.x * blockDim.x + threadIdx.x;
    int e    = gtid >> 5;
    if (e >= TOT_EDGES) return;
    int lane = threadIdx.x & 31;

    int s, d;
    if (e < N_EDGES) {
        s = (int)ei[e];
        d = (int)ei[(size_t)N_EDGES + e];
        if (lane == 0) __builtin_prefetch(ei + e + 1024, 0, 0);  // global_prefetch_b8
    } else {
        s = d = e - N_EDGES;                   // self-loop
    }
    float nrm = dinv[s] * dinv[d];

    float4 h4 = *(const float4*)(H + (size_t)s * 128 + lane * 4);
    float* out = X + (size_t)d * 128 + lane * 4;
    atomicAdd(out + 0, h4.x * nrm);
    atomicAdd(out + 1, h4.y * nrm);
    atomicAdd(out + 2, h4.z * nrm);
    atomicAdd(out + 3, h4.w * nrm);
}

// ---------------- bias + ELU, in place ----------------
__global__ void bias_elu_kernel(float* __restrict__ X, const float* __restrict__ b, int n4) {
    int i = blockIdx.x * blockDim.x + threadIdx.x;
    if (i >= n4) return;
    int i4 = i * 4;
    int f0 = i4 & 127;
    float4 v  = *(float4*)(X + i4);
    float4 bv = *(const float4*)(b + f0);
    float x0 = v.x + bv.x, x1 = v.y + bv.y, x2 = v.z + bv.z, x3 = v.w + bv.w;
    v.x = x0 > 0.f ? x0 : __expf(x0) - 1.f;
    v.y = x1 > 0.f ? x1 : __expf(x1) - 1.f;
    v.z = x2 > 0.f ? x2 : __expf(x2) - 1.f;
    v.w = x3 > 0.f ? x3 : __expf(x3) - 1.f;
    *(float4*)(X + i4) = v;
}

// ---------------- global max pool ----------------
__global__ void pool_init_kernel(float* pool) {
    int i = blockIdx.x * blockDim.x + threadIdx.x;
    if (i < N_GRAPHS * N_HID) pool[i] = -3.0e38f;
}

__global__ void pool_max_kernel(const float* __restrict__ X,
                                const long long* __restrict__ batch,
                                float* __restrict__ pool) {
    int i = blockIdx.x * blockDim.x + threadIdx.x;
    int n4 = (N_NODES * N_HID) / 4;
    if (i >= n4) return;
    int i4   = i * 4;
    int node = i4 >> 7;
    int f0   = i4 & 127;
    int g    = (int)batch[node];
    float4 v = *(const float4*)(X + i4);
    float* p = pool + (size_t)g * 128 + f0;
    __hip_atomic_fetch_max(p + 0, v.x, __ATOMIC_RELAXED, __HIP_MEMORY_SCOPE_AGENT);
    __hip_atomic_fetch_max(p + 1, v.y, __ATOMIC_RELAXED, __HIP_MEMORY_SCOPE_AGENT);
    __hip_atomic_fetch_max(p + 2, v.z, __ATOMIC_RELAXED, __HIP_MEMORY_SCOPE_AGENT);
    __hip_atomic_fetch_max(p + 3, v.w, __ATOMIC_RELAXED, __HIP_MEMORY_SCOPE_AGENT);
}

// ---------------- classifier + softmax: one wave per graph ----------------
__global__ void classify_kernel(const float* __restrict__ pool,
                                const float* __restrict__ Wc,
                                const float* __restrict__ bc,
                                float* __restrict__ out) {
    __shared__ float logits[N_CLASSES];
    int row  = blockIdx.x;
    int lane = threadIdx.x;
    if (lane < N_CLASSES) {
        float s = bc[lane];
        const float* g = pool + (size_t)row * 128;
        #pragma unroll 8
        for (int k = 0; k < 128; ++k) s += g[k] * Wc[k * N_CLASSES + lane];
        logits[lane] = s;
    }
    __syncthreads();
    if (lane < N_CLASSES) {
        float m = logits[0];
        #pragma unroll
        for (int c = 1; c < N_CLASSES; ++c) m = fmaxf(m, logits[c]);
        float denom = 0.f;
        #pragma unroll
        for (int c = 0; c < N_CLASSES; ++c) denom += __expf(logits[c] - m);
        out[(size_t)row * N_CLASSES + lane] = __expf(logits[lane] - m) / denom;
    }
}

// ---------------- launch ----------------
extern "C" void kernel_launch(void* const* d_in, const int* in_sizes, int n_in,
                              void* d_out, int out_size, void* d_ws, size_t ws_size,
                              hipStream_t stream) {
    const float*     x_in  = (const float*)d_in[0];
    const long long* ei    = (const long long*)d_in[1];   // [2, E] int64
    const long long* batch = (const long long*)d_in[2];   // [N] int64
    const float*     Ws    = (const float*)d_in[3];       // [3,128,128]
    const float*     bs    = (const float*)d_in[4];       // [3,128]
    const float*     Wc    = (const float*)d_in[5];       // [128,10]
    const float*     bc    = (const float*)d_in[6];       // [10]
    float*           out   = (float*)d_out;               // [256,10]

    // workspace layout: X (accum / activations), H (transform output),
    // dinv, pool.  Total ~103 MB.
    float* X    = (float*)d_ws;
    float* H    = X + (size_t)N_NODES * N_HID;
    float* dinv = H + (size_t)N_NODES * N_HID;
    float* pool = dinv + N_NODES;

    const int nodes_blk  = (N_NODES + 255) / 256;
    const int edges_blk  = (N_EDGES + 255) / 256;
    const int feat4      = (N_NODES * N_HID) / 4;          // 3.2M
    const int feat4_blk  = (feat4 + 255) / 256;
    const int scat_blk   = (TOT_EDGES * 32 + 255) / 256;   // 1 wave per edge
    const int gemm_blk   = (M_TILES + 3) / 4;

    // degree / normalization
    deg_init_kernel<<<nodes_blk, 256, 0, stream>>>(dinv);
    deg_accum_kernel<<<edges_blk, 256, 0, stream>>>(ei, dinv);
    deg_finalize_kernel<<<nodes_blk, 256, 0, stream>>>(dinv);

    pool_init_kernel<<<(N_GRAPHS * N_HID + 255) / 256, 256, 0, stream>>>(pool);

    for (int l = 0; l < N_LAYERS; ++l) {
        const float* xin = (l == 0) ? x_in : X;
        gemm_wmma_kernel<<<gemm_blk, 128, 0, stream>>>(xin, Ws + (size_t)l * N_HID * N_HID, H);
        zero_kernel<<<feat4_blk, 256, 0, stream>>>((float4*)X, feat4);
        scatter_kernel<<<scat_blk, 256, 0, stream>>>(ei, dinv, H, X);
        bias_elu_kernel<<<feat4_blk, 256, 0, stream>>>(X, bs + (size_t)l * N_HID, feat4);
    }

    pool_max_kernel<<<feat4_blk, 256, 0, stream>>>(X, batch, pool);
    classify_kernel<<<N_GRAPHS, 32, 0, stream>>>(pool, Wc, bc, out);
}
